// SRCapsNet_89644557402730
// MI455X (gfx1250) — matrix operand order repaired and checked by reading, not compiled, not run on hardware
//
#include <hip/hip_runtime.h>
#include <hip/hip_bf16.h>
#include <math.h>

// ---------------- problem constants ----------------
constexpr int NB   = 32;    // batch
constexpr int NTOK = 64;    // L
constexpr int NT   = 65;    // L + cls
constexpr int ND   = 256;   // D
constexpr int NH   = 8;     // heads
constexpr int NDH  = 32;    // head dim
constexpr int NDFF = 2048;
constexpr int NS   = 16;    // slots
constexpr int NI   = 16;    // intents
constexpr int NDSEM= 128;
constexpr int NLAY = 2;

// ---------------- WMMA plumbing (CDNA5, wave32) ----------------
typedef __attribute__((ext_vector_type(16))) __bf16 v16bf;
typedef __attribute__((ext_vector_type(8)))  float  v8f;

__device__ __forceinline__ v8f wmma_bf16(v16bf a, v16bf b, v8f c) {
  // v_wmma_f32_16x16x32_bf16  D = A(16x32) * B(32x16) + C
  return __builtin_amdgcn_wmma_f32_16x16x32_bf16(false, a, false, b, (short)0, c,
                                                 false, false);
}

// A fragment: 16x32 bf16. lane&15 = M row; lane>>4 selects K half.
// element e -> K = kb + half*16 + e   (contiguous, 64B)
__device__ __forceinline__ v16bf load_a_frag(const float* __restrict__ arow,
                                             int kb, int half, float scale) {
  const float4* p0 = (const float4*)(arow + kb + half * 16);
  const float4* p1 = (const float4*)(arow + kb + half * 16 + 8);
  float4 x0 = p0[0], x1 = p0[1], y0 = p1[0], y1 = p1[1];
  v16bf a;
  a[0]  = (__bf16)(x0.x * scale); a[1]  = (__bf16)(x0.y * scale);
  a[2]  = (__bf16)(x0.z * scale); a[3]  = (__bf16)(x0.w * scale);
  a[4]  = (__bf16)(x1.x * scale); a[5]  = (__bf16)(x1.y * scale);
  a[6]  = (__bf16)(x1.z * scale); a[7]  = (__bf16)(x1.w * scale);
  a[8]  = (__bf16)(y0.x * scale); a[9]  = (__bf16)(y0.y * scale);
  a[10] = (__bf16)(y0.z * scale); a[11] = (__bf16)(y0.w * scale);
  a[12] = (__bf16)(y1.x * scale); a[13] = (__bf16)(y1.y * scale);
  a[14] = (__bf16)(y1.z * scale); a[15] = (__bf16)(y1.w * scale);
  return a;
}

// B fragment: 32x16 bf16. lane&15 = N col; element e -> K = kb + half*16 + e.
// Weight stored (N,K) row-major -> one contiguous 64B run per lane.
__device__ __forceinline__ v16bf load_b_frag_nk(const float* __restrict__ wrow,
                                                int kb, int half) {
  const float4* p = (const float4*)(wrow + kb + half * 16);
  float4 x0 = p[0], x1 = p[1], x2 = p[2], x3 = p[3];
  v16bf b;
  b[0]  = (__bf16)x0.x; b[1]  = (__bf16)x0.y; b[2]  = (__bf16)x0.z; b[3]  = (__bf16)x0.w;
  b[4]  = (__bf16)x1.x; b[5]  = (__bf16)x1.y; b[6]  = (__bf16)x1.z; b[7]  = (__bf16)x1.w;
  b[8]  = (__bf16)x2.x; b[9]  = (__bf16)x2.y; b[10] = (__bf16)x2.z; b[11] = (__bf16)x2.w;
  b[12] = (__bf16)x3.x; b[13] = (__bf16)x3.y; b[14] = (__bf16)x3.z; b[15] = (__bf16)x3.w;
  return b;
}

// ---------------- generic GEMM: C = act(A @ W^T + bias) ----------------
// A: (M,K) f32 row-major.  W: (N,K) f32 row-major.  One wave = 32x32 C tile:
// 2 A-frags x 2 B-frags -> 4 WMMAs per K step (halves A and B traffic).
__global__ void wmma_gemm_kernel(const float* __restrict__ A,
                                 const float* __restrict__ W,
                                 const float* __restrict__ bias,
                                 float* __restrict__ C,
                                 int M, int N, int K, int relu) {
  const int lane = threadIdx.x;
  const int half = lane >> 4, q = lane & 15;
  const int m0 = blockIdx.y * 32;
  const int n0 = blockIdx.x * 32;
  const float* arow0 = A + (size_t)(m0 + q) * K;
  const float* arow1 = A + (size_t)(m0 + 16 + q) * K;
  const float* wrow0 = W + (size_t)(n0 + q) * K;
  const float* wrow1 = W + (size_t)(n0 + 16 + q) * K;
  v8f acc00 = {}, acc01 = {}, acc10 = {}, acc11 = {};
  for (int kb = 0; kb < K; kb += 32) {
    __builtin_prefetch(wrow0 + kb + 128, 0, 1);   // global_prefetch next tiles
    __builtin_prefetch(wrow1 + kb + 128, 0, 1);
    v16bf a0 = load_a_frag(arow0, kb, half, 1.0f);
    v16bf a1 = load_a_frag(arow1, kb, half, 1.0f);
    v16bf b0 = load_b_frag_nk(wrow0, kb, half);
    v16bf b1 = load_b_frag_nk(wrow1, kb, half);
    acc00 = wmma_bf16(a0, b0, acc00);
    acc01 = wmma_bf16(a0, b1, acc01);
    acc10 = wmma_bf16(a1, b0, acc10);
    acc11 = wmma_bf16(a1, b1, acc11);
  }
  const int na = n0 + q, nb = n0 + 16 + q;
  const float bva = bias[na], bvb = bias[nb];
#pragma unroll
  for (int r = 0; r < 8; ++r) {
    int ma = m0 + r + half * 8;                    // C layout §7.12.2
    int mb = ma + 16;
    float v00 = acc00[r] + bva, v01 = acc01[r] + bvb;
    float v10 = acc10[r] + bva, v11 = acc11[r] + bvb;
    if (relu) {
      v00 = fmaxf(v00, 0.0f); v01 = fmaxf(v01, 0.0f);
      v10 = fmaxf(v10, 0.0f); v11 = fmaxf(v11, 0.0f);
    }
    C[(size_t)ma * N + na] = v00;
    C[(size_t)ma * N + nb] = v01;
    C[(size_t)mb * N + na] = v10;
    C[(size_t)mb * N + nb] = v11;
  }
}

// (K,N) -> (N,K) one-time transpose for W_feat (tiny: 256x256)
__global__ void transpose_kernel(const float* __restrict__ Win,
                                 float* __restrict__ Wout, int K, int N) {
  __shared__ float tile[16][17];
  int k0 = blockIdx.x * 16, n0 = blockIdx.y * 16;
  int tx = threadIdx.x & 15, ty = threadIdx.x >> 4;   // 256 threads
  tile[ty][tx] = Win[(size_t)(k0 + ty) * N + n0 + tx];
  __syncthreads();
  Wout[(size_t)(n0 + ty) * K + k0 + tx] = tile[tx][ty];
}

// ---------------- small utility kernels ----------------
__device__ __forceinline__ float block_sum(float v, float* sh, int n) {
  int t = threadIdx.x;
  sh[t] = v; __syncthreads();
  for (int off = n >> 1; off > 0; off >>= 1) {
    if (t < off) sh[t] += sh[t + off];
    __syncthreads();
  }
  float r = sh[0]; __syncthreads();
  return r;
}

__global__ void concat_kernel(const float* __restrict__ dense,
                              const float* __restrict__ sparse,
                              float* __restrict__ cat) {
  int idx = blockIdx.x * 256 + threadIdx.x;      // B*L*256 total
  int c = idx & 255, bl = idx >> 8;
  cat[idx] = (c < 128) ? dense[bl * 128 + c] : sparse[bl * 128 + (c - 128)];
}

// tok(B*L,D) -> x(B,T,D) with cls = sum over L in row T-1
__global__ void assemble_x_kernel(const float* __restrict__ tok,
                                  float* __restrict__ x) {
  int b = blockIdx.x, d = threadIdx.x;           // 256 threads
  float s = 0.0f;
  for (int l = 0; l < NTOK; ++l) {
    float v = tok[((size_t)(b * NTOK + l)) * ND + d];
    x[((size_t)(b * NT + l)) * ND + d] = v;
    s += v;
  }
  x[((size_t)(b * NT + NTOK)) * ND + d] = s;
}

// x[row] = LN(x[row] + r[row]) * g + b     (in-place, row-local)
__global__ void add_ln_kernel(float* __restrict__ x, const float* __restrict__ r,
                              const float* __restrict__ g, const float* __restrict__ bt) {
  __shared__ float sh[ND];
  int row = blockIdx.x, t = threadIdx.x;
  size_t off = (size_t)row * ND + t;
  float v = x[off] + r[off];
  float mean = block_sum(v, sh, ND) * (1.0f / ND);
  float d = v - mean;
  float var = block_sum(d * d, sh, ND) * (1.0f / ND);
  x[off] = d * rsqrtf(var + 1e-5f) * g[t] + bt[t];
}

// one-pass online-softmax attention, one (b,h) per block, one query per thread
__global__ void attention_kernel(const float* __restrict__ qkv,
                                 float* __restrict__ attn) {
  __shared__ float Ks[NT][NDH];
  __shared__ float Vs[NT][NDH];
  int b = blockIdx.x / NH, h = blockIdx.x % NH;
  int tid = threadIdx.x;                          // 128 threads
  for (int idx = tid; idx < NT * NDH; idx += 128) {
    int t = idx >> 5, d = idx & 31;
    const float* row = qkv + (size_t)(b * NT + t) * (3 * ND);
    Ks[t][d] = row[ND + h * NDH + d];
    Vs[t][d] = row[2 * ND + h * NDH + d];
  }
  __syncthreads();
  if (tid < NT) {
    const float* qrow = qkv + (size_t)(b * NT + tid) * (3 * ND) + h * NDH;
    const float scl = 0.17677669529663687f;       // 1/sqrt(32)
    float q[NDH];
#pragma unroll
    for (int d = 0; d < NDH; ++d) q[d] = qrow[d] * scl;
    float m = -1e30f, ssum = 0.0f, acc[NDH];
#pragma unroll
    for (int d = 0; d < NDH; ++d) acc[d] = 0.0f;
    for (int t = 0; t < NT; ++t) {
      float s = 0.0f;
#pragma unroll
      for (int d = 0; d < NDH; ++d) s += q[d] * Ks[t][d];
      if (s > m) {
        float c = __expf(m - s);
        ssum *= c;
#pragma unroll
        for (int d = 0; d < NDH; ++d) acc[d] *= c;
        m = s;
      }
      float w = __expf(s - m);
      ssum += w;
#pragma unroll
      for (int d = 0; d < NDH; ++d) acc[d] += w * Vs[t][d];
    }
    float inv = 1.0f / ssum;
    float* o = attn + (size_t)(b * NT + tid) * ND + h * NDH;
#pragma unroll
    for (int d = 0; d < NDH; ++d) o[d] = acc[d] * inv;
  }
}

// per (b,l) one wave: a_words = sigmoid(w·w_act), c_ws = softmax_s(w·w_route)
__global__ void route_ws_kernel(const float* __restrict__ x,
                                const float* __restrict__ w_route,
                                const float* __restrict__ w_act,
                                float* __restrict__ c_ws_out,
                                float* __restrict__ a_words) {
  __shared__ float dots[NS + 1];
  int bl = blockIdx.x, b = bl / NTOK, l = bl % NTOK;
  int lane = threadIdx.x;
  const float* wrow = x + (size_t)(b * NT + l) * ND;
  for (int s = 0; s <= NS; ++s) {
    const float* w = (s < NS) ? (w_route + (size_t)(l * NS + s) * ND) : w_act;
    float p = 0.0f;
    for (int j = lane; j < ND; j += 32) p += wrow[j] * w[j];
    for (int off = 16; off; off >>= 1) p += __shfl_xor(p, off, 32);
    if (lane == 0) dots[s] = p;
  }
  __syncthreads();
  if (lane == 0) {
    float mx = dots[0];
    for (int s = 1; s < NS; ++s) mx = fmaxf(mx, dots[s]);
    float sum = 0.0f, e[NS];
    for (int s = 0; s < NS; ++s) { e[s] = __expf(dots[s] - mx); sum += e[s]; }
    float inv = 1.0f / sum;
    for (int s = 0; s < NS; ++s) c_ws_out[(size_t)bl * NS + s] = e[s] * inv;
    a_words[bl] = 1.0f / (1.0f + __expf(-dots[NS]));
  }
}

__global__ void slot_stats_kernel(const float* __restrict__ c_ws,
                                  const float* __restrict__ a_words,
                                  float* __restrict__ a_num,
                                  float* __restrict__ a_slots) {
  __shared__ float sh[NTOK];
  int b = blockIdx.x, l = threadIdx.x;            // 64 threads
  float aw = a_words[b * NTOK + l];
  sh[l] = aw; __syncthreads();
  for (int off = 32; off; off >>= 1) { if (l < off) sh[l] += sh[l + off]; __syncthreads(); }
  float tot = sh[0]; __syncthreads();
  for (int s = 0; s < NS; ++s) {
    sh[l] = aw * c_ws[(size_t)(b * NTOK + l) * NS + s]; __syncthreads();
    for (int off = 32; off; off >>= 1) { if (l < off) sh[l] += sh[l + off]; __syncthreads(); }
    if (l == 0) { a_num[b * NS + s] = sh[0]; a_slots[b * NS + s] = sh[0] / tot; }
    __syncthreads();
  }
}

// u_slots[b,s,i] = (1/a_num) * sum_{l,j} (c_ws*a_words*words)[b,l,j] * Wp[l,s,i,j]
// Flat-K GEMM, K = L*D = 16384.  One wave covers BOTH 16-row M halves so the
// 268MB w_pose_ws tensor is streamed exactly once; scales hoisted per token.
__global__ void pose_ws_kernel(const float* __restrict__ x,
                               const float* __restrict__ c_ws,
                               const float* __restrict__ a_words,
                               const float* __restrict__ w_pose,
                               const float* __restrict__ a_num,
                               float* __restrict__ u_slots) {
  const int lane = threadIdx.x;
  const int half = lane >> 4, q = lane & 15;
  const int ntile = blockIdx.x;                   // 16 tiles of i
  const int s = blockIdx.y;                       // slot
  const int i = ntile * 16 + q;
  const int b0 = q, b1 = 16 + q;
  v8f acc0 = {}, acc1 = {};
  for (int l = 0; l < NTOK; ++l) {
    const float sc0 = c_ws[(size_t)(b0 * NTOK + l) * NS + s] * a_words[b0 * NTOK + l];
    const float sc1 = c_ws[(size_t)(b1 * NTOK + l) * NS + s] * a_words[b1 * NTOK + l];
    const float* ar0 = x + (size_t)(b0 * NT + l) * ND;
    const float* ar1 = x + (size_t)(b1 * NT + l) * ND;
    const float* wrow = w_pose + (((size_t)(l * NS + s)) * ND + i) * ND;
#pragma unroll
    for (int jb = 0; jb < ND; jb += 32) {
      __builtin_prefetch(wrow + jb + 128, 0, 1);
      v16bf bf = load_b_frag_nk(wrow, jb, half);
      acc0 = wmma_bf16(load_a_frag(ar0, jb, half, sc0), bf, acc0);
      acc1 = wmma_bf16(load_a_frag(ar1, jb, half, sc1), bf, acc1);
    }
  }
#pragma unroll
  for (int r = 0; r < 8; ++r) {
    int m0 = r + half * 8, m1 = 16 + m0;
    u_slots[((size_t)(m0 * NS + s)) * ND + i] = acc0[r] / a_num[m0 * NS + s];
    u_slots[((size_t)(m1 * NS + s)) * ND + i] = acc1[r] / a_num[m1 * NS + s];
  }
}

// per b: c_si softmax over i, an, a_intents
__global__ void route_si_kernel(const float* __restrict__ u_slots,
                                const float* __restrict__ w_route_si,
                                const float* __restrict__ a_slots,
                                float* __restrict__ c_si,
                                float* __restrict__ an,
                                float* __restrict__ a_int) {
  __shared__ float sh[256];
  int b = blockIdx.x, t = threadIdx.x;            // 256 threads = (s,i)
  int s = t >> 4, i = t & 15;
  const float* u = u_slots + (size_t)(b * NS + s) * ND;
  const float* w = w_route_si + (size_t)(s * NI + i) * ND;
  float dot = 0.0f;
  for (int j = 0; j < ND; ++j) dot += u[j] * w[j];
  sh[t] = dot; __syncthreads();
  float mx = sh[s * 16];
  for (int k = 1; k < 16; ++k) mx = fmaxf(mx, sh[s * 16 + k]);
  float e = __expf(dot - mx);
  __syncthreads(); sh[t] = e; __syncthreads();
  float sum = 0.0f;
  for (int k = 0; k < 16; ++k) sum += sh[s * 16 + k];
  float c = e / sum;
  c_si[(size_t)(b * NS + s) * NI + i] = c;
  __syncthreads(); sh[t] = c * a_slots[b * NS + s]; __syncthreads();
  if (s == 0) {
    float a = 0.0f;
    for (int k = 0; k < NS; ++k) a += sh[k * 16 + i];
    float asum = 0.0f;
    for (int k = 0; k < NS; ++k) asum += a_slots[b * NS + k];
    an[b * NI + i] = a;
    a_int[b * NI + i] = a / (asum + 1e-8f);
  }
}

// u_intents[b,i,k] = (1/(an+eps)) * sum_{s,kk} (c_si*a_slots*u_slots)[b,s,kk] * Wsi[s,i,k,kk]
// Flat-K GEMM, K = S*D = 4096, both M halves per wave, scales hoisted per slot.
__global__ void pose_si_kernel(const float* __restrict__ u_slots,
                               const float* __restrict__ c_si,
                               const float* __restrict__ a_slots,
                               const float* __restrict__ w_pose_si,
                               const float* __restrict__ an,
                               float* __restrict__ u_int) {
  const int lane = threadIdx.x;
  const int half = lane >> 4, q = lane & 15;
  const int ntile = blockIdx.x;                   // 16 tiles of k
  const int i = blockIdx.y;                       // intent
  const int kcol = ntile * 16 + q;
  const int b0 = q, b1 = 16 + q;
  v8f acc0 = {}, acc1 = {};
  for (int s = 0; s < NS; ++s) {
    const float sc0 = c_si[(size_t)(b0 * NS + s) * NI + i] * a_slots[b0 * NS + s];
    const float sc1 = c_si[(size_t)(b1 * NS + s) * NI + i] * a_slots[b1 * NS + s];
    const float* ar0 = u_slots + (size_t)(b0 * NS + s) * ND;
    const float* ar1 = u_slots + (size_t)(b1 * NS + s) * ND;
    const float* wrow = w_pose_si + (((size_t)(s * NI + i)) * ND + kcol) * ND;
#pragma unroll
    for (int kk = 0; kk < ND; kk += 32) {
      __builtin_prefetch(wrow + kk + 128, 0, 1);
      v16bf bf = load_b_frag_nk(wrow, kk, half);
      acc0 = wmma_bf16(load_a_frag(ar0, kk, half, sc0), bf, acc0);
      acc1 = wmma_bf16(load_a_frag(ar1, kk, half, sc1), bf, acc1);
    }
  }
#pragma unroll
  for (int r = 0; r < 8; ++r) {
    int m0 = r + half * 8, m1 = 16 + m0;
    u_int[((size_t)(m0 * NI + i)) * ND + kcol] = acc0[r] / (an[m0 * NI + i] + 1e-8f);
    u_int[((size_t)(m1 * NI + i)) * ND + kcol] = acc1[r] / (an[m1 * NI + i] + 1e-8f);
  }
}

// argmax + classifier + layernorm -> intents
__global__ void final_kernel(const float* __restrict__ x,
                             const float* __restrict__ a_int,
                             const float* __restrict__ u_int,
                             const float* __restrict__ cls_w,
                             const float* __restrict__ cls_b,
                             const float* __restrict__ g,
                             const float* __restrict__ bt,
                             float* __restrict__ out) {
  __shared__ float sh[NDSEM];
  __shared__ int idx_s;
  int b = blockIdx.x, t = threadIdx.x;            // 128 threads
  if (t == 0) {
    int best = 0; float bv = a_int[b * NI];
    for (int i = 1; i < NI; ++i) {
      float v = a_int[b * NI + i];
      if (v > bv) { bv = v; best = i; }
    }
    idx_s = best;
  }
  __syncthreads();
  const float* cls_tok = x + (size_t)(b * NT + NTOK) * ND;
  const float* pose = u_int + (size_t)(b * NI + idx_s) * ND;
  const float* w = cls_w + (size_t)t * (2 * ND);
  float acc = cls_b[t];
  for (int j = 0; j < ND; ++j) acc += cls_tok[j] * w[j];
  for (int j = 0; j < ND; ++j) acc += pose[j] * w[ND + j];
  float mean = block_sum(acc, sh, NDSEM) * (1.0f / NDSEM);
  float d = acc - mean;
  float var = block_sum(d * d, sh, NDSEM) * (1.0f / NDSEM);
  out[b * NDSEM + t] = d * rsqrtf(var + 1e-5f) * g[t] + bt[t];
}

// ---------------- host launch ----------------
extern "C" void kernel_launch(void* const* d_in, const int* in_sizes, int n_in,
                              void* d_out, int out_size, void* d_ws, size_t ws_size,
                              hipStream_t stream) {
  const float* dense      = (const float*)d_in[0];
  const float* sparse     = (const float*)d_in[1];
  const float* W_feat     = (const float*)d_in[2];
  const float* b_feat     = (const float*)d_in[3];
  const float* in_proj_w  = (const float*)d_in[4];
  const float* in_proj_b  = (const float*)d_in[5];
  const float* out_w      = (const float*)d_in[6];
  const float* out_b      = (const float*)d_in[7];
  const float* ln1_g      = (const float*)d_in[8];
  const float* ln1_b      = (const float*)d_in[9];
  const float* ln2_g      = (const float*)d_in[10];
  const float* ln2_b      = (const float*)d_in[11];
  const float* ff1_w      = (const float*)d_in[12];
  const float* ff1_b      = (const float*)d_in[13];
  const float* ff2_w      = (const float*)d_in[14];
  const float* ff2_b      = (const float*)d_in[15];
  const float* w_act      = (const float*)d_in[16];
  const float* w_route_ws = (const float*)d_in[17];
  const float* w_pose_ws  = (const float*)d_in[18];
  const float* w_route_si = (const float*)d_in[19];
  const float* w_pose_si  = (const float*)d_in[20];
  const float* cls_w      = (const float*)d_in[21];
  const float* cls_b      = (const float*)d_in[22];
  const float* inorm_g    = (const float*)d_in[23];
  const float* inorm_b    = (const float*)d_in[24];

  float* ws = (float*)d_ws;
  size_t o = 0;
  float* cat     = ws + o; o += (size_t)NB * NTOK * ND;
  float* Wt_feat = ws + o; o += (size_t)ND * ND;         // (N,K) transposed W_feat
  float* tok     = ws + o; o += (size_t)NB * NTOK * ND;
  float* x       = ws + o; o += (size_t)NB * NT * ND;
  float* qkv     = ws + o; o += (size_t)NB * NT * 3 * ND;
  float* attnO   = ws + o; o += (size_t)NB * NT * ND;
  float* proj    = ws + o; o += (size_t)NB * NT * ND;
  float* ff1     = ws + o; o += (size_t)NB * NT * NDFF;
  float* ff2     = ws + o; o += (size_t)NB * NT * ND;
  float* a_words = ws + o; o += (size_t)NB * NTOK;
  float* a_num   = ws + o; o += (size_t)NB * NS;
  float* a_slots = ws + o; o += (size_t)NB * NS;
  float* u_slots = ws + o; o += (size_t)NB * NS * ND;
  float* c_si    = ws + o; o += (size_t)NB * NS * NI;
  float* an      = ws + o; o += (size_t)NB * NI;
  float* a_int   = ws + o; o += (size_t)NB * NI;
  float* u_int   = ws + o; o += (size_t)NB * NI * ND;

  float* outF = (float*)d_out;                 // c_ws (B*L*S) then intents (B*128)

  const int Mtok = NB * NTOK;                  // 2048
  const int Mx   = NB * NT;                    // 2080

  // 1) features
  concat_kernel<<<(NB * NTOK * ND) / 256, 256, 0, stream>>>(dense, sparse, cat);
  transpose_kernel<<<dim3(ND / 16, ND / 16), 256, 0, stream>>>(W_feat, Wt_feat, ND, ND);
  wmma_gemm_kernel<<<dim3(ND / 32, Mtok / 32), 32, 0, stream>>>(
      cat, Wt_feat, b_feat, tok, Mtok, ND, ND, 0);
  assemble_x_kernel<<<NB, ND, 0, stream>>>(tok, x);

  // 2) encoder layers
  for (int i = 0; i < NLAY; ++i) {
    const float* wq = in_proj_w + (size_t)i * 3 * ND * ND;
    const float* bq = in_proj_b + (size_t)i * 3 * ND;
    const float* wo = out_w + (size_t)i * ND * ND;
    const float* bo = out_b + (size_t)i * ND;
    const float* w1 = ff1_w + (size_t)i * NDFF * ND;
    const float* b1 = ff1_b + (size_t)i * NDFF;
    const float* w2 = ff2_w + (size_t)i * ND * NDFF;
    const float* b2 = ff2_b + (size_t)i * ND;

    wmma_gemm_kernel<<<dim3(3 * ND / 32, Mx / 32), 32, 0, stream>>>(
        x, wq, bq, qkv, Mx, 3 * ND, ND, 0);
    attention_kernel<<<NB * NH, 128, 0, stream>>>(qkv, attnO);
    wmma_gemm_kernel<<<dim3(ND / 32, Mx / 32), 32, 0, stream>>>(
        attnO, wo, bo, proj, Mx, ND, ND, 0);
    add_ln_kernel<<<Mx, ND, 0, stream>>>(x, proj,
        ln1_g + (size_t)i * ND, ln1_b + (size_t)i * ND);
    wmma_gemm_kernel<<<dim3(NDFF / 32, Mx / 32), 32, 0, stream>>>(
        x, w1, b1, ff1, Mx, NDFF, ND, /*relu=*/1);
    wmma_gemm_kernel<<<dim3(ND / 32, Mx / 32), 32, 0, stream>>>(
        ff1, w2, b2, ff2, Mx, ND, NDFF, 0);
    add_ln_kernel<<<Mx, ND, 0, stream>>>(x, ff2,
        ln2_g + (size_t)i * ND, ln2_b + (size_t)i * ND);
  }

  // 3) word -> slot routing  (c_ws goes straight to d_out)
  route_ws_kernel<<<NB * NTOK, 32, 0, stream>>>(x, w_route_ws, w_act, outF, a_words);
  slot_stats_kernel<<<NB, NTOK, 0, stream>>>(outF, a_words, a_num, a_slots);
  pose_ws_kernel<<<dim3(ND / 16, NS), 32, 0, stream>>>(
      x, outF, a_words, w_pose_ws, a_num, u_slots);

  // 4) slot -> intent routing
  route_si_kernel<<<NB, 256, 0, stream>>>(u_slots, w_route_si, a_slots, c_si, an, a_int);
  pose_si_kernel<<<dim3(ND / 16, NI), 32, 0, stream>>>(
      u_slots, c_si, a_slots, w_pose_si, an, u_int);

  // 5) classifier + layernorm -> intents
  final_kernel<<<NB, NDSEM, 0, stream>>>(x, a_int, u_int, cls_w, cls_b,
                                         inorm_g, inorm_b,
                                         outF + (size_t)NB * NTOK * NS);
}
